// SimCLRLoss_31542239822525
// MI455X (gfx1250) — matrix operand order repaired and checked by reading, not compiled, not run on hardware
//
#include <hip/hip_runtime.h>
#include <hip/hip_bf16.h>

typedef __attribute__((ext_vector_type(16))) _Float16 v16h;
typedef __attribute__((ext_vector_type(8)))  _Float16 v8h;
typedef __attribute__((ext_vector_type(8)))  float    v8f;

#define NPAIR 4096   // N
#define TWO_N 8192   // 2N rows of z
#define DDIM  256    // feature dim
// s = sqrt( (1/T) * log2(e) ) = sqrt(2 * 1.4426950408889634)
// -> dot(zn_a, zn_b) = sim * log2(e), so exp(sim) == exp2(dot): bare v_exp_f32.
#define SCALE_L2E 1.6986436f
#define LN2       0.69314718056f

// ---------------------------------------------------------------------------
// Kernel 1: row L2-normalize, scale by sqrt(log2e/T), convert to f16.
// One block per row, one thread per element (D == 256). Block 0 zeroes out[0].
// ---------------------------------------------------------------------------
__global__ __launch_bounds__(256)
void simclr_norm_f16(const float* __restrict__ zi, const float* __restrict__ zj,
                     _Float16* __restrict__ zn, float* __restrict__ out) {
    int row = blockIdx.x;          // 0..8191
    int t   = threadIdx.x;         // 0..255
    const float* src = (row < NPAIR) ? (zi + (size_t)row * DDIM)
                                     : (zj + (size_t)(row - NPAIR) * DDIM);
    float x  = src[t];
    float ss = x * x;
    #pragma unroll
    for (int off = 16; off >= 1; off >>= 1)
        ss += __shfl_xor(ss, off, 32);

    __shared__ float red[8];
    if ((t & 31) == 0) red[t >> 5] = ss;
    __syncthreads();
    if (t == 0) {
        float tot = 0.f;
        #pragma unroll
        for (int i = 0; i < 8; ++i) tot += red[i];
        red[0] = rsqrtf(tot) * SCALE_L2E;
    }
    __syncthreads();
    float scale = red[0];
    zn[(size_t)row * DDIM + t] = (_Float16)(x * scale);

    if (row == 0 && t == 0) out[0] = 0.0f;   // init accumulator for kernel 2
}

// ---------------------------------------------------------------------------
// WMMA operand loaders from row-major zn[8192][256] (f16), wave32 layouts.
// A 16x32 (MxK): lane half h, row M = lane&15.
//   VGPR0-3: K = kbase + 8h + [0..8);  VGPR4-7: K = kbase + 16 + 8h + [0..8)
// B 32x16 (KxN): col N = lane&15 -> row (colbase + N) of zn.
//   VGPR0-3: K = kbase + 8h + [0..8);  VGPR4-7: K = kbase + 16 + 8h + [0..8)
// ---------------------------------------------------------------------------
__device__ __forceinline__ v16h load_A(const _Float16* __restrict__ zn,
                                       int row, int kbase, int h) {
    const _Float16* p = zn + (size_t)row * DDIM + kbase + h * 8;
    union { v16h v; v8h q[2]; } u;
    u.q[0] = *(const v8h*)(p);        // global_load_b128
    u.q[1] = *(const v8h*)(p + 16);   // global_load_b128
    return u.v;
}

__device__ __forceinline__ void load_Btile(v16h B[8], const _Float16* __restrict__ zn,
                                           int brow, int h) {
    #pragma unroll
    for (int kt = 0; kt < 8; ++kt) {
        const _Float16* p = zn + (size_t)brow * DDIM + kt * 32 + h * 8;
        union { v16h v; v8h q[2]; } u;
        u.q[0] = *(const v8h*)(p);
        u.q[1] = *(const v8h*)(p + 16);
        B[kt] = u.v;
    }
}

__device__ __forceinline__ v8f gemm_tile(const v16h A[8], const v16h B[8]) {
    v8f c = {};
    #pragma unroll
    for (int kt = 0; kt < 8; ++kt)
        c = __builtin_amdgcn_wmma_f32_16x16x32_f16(
                false, A[kt], false, B[kt], (short)0, c, false, false);
    return c;
}

// Epilogue: special tiles selected by WAVE-UNIFORM compares -> s_cbranch,
// zero exec-mask churn in the steady state (510 of 512 tiles take fast path).
__device__ __forceinline__ void epilogue(const v8f& c, int colbase, int rowbase,
                                         int n, int mh, v8f& rowacc,
                                         float* __restrict__ s_pos) {
    if (colbase == rowbase) {                       // diagonal tile: mask exp
        #pragma unroll
        for (int r = 0; r < 8; ++r) {
            float e = __builtin_amdgcn_exp2f(c[r]); // v_exp_f32
            if (n == r + mh) e = 0.0f;
            rowacc[r] += e;
        }
    } else if (colbase == (rowbase ^ NPAIR)) {      // positive-pair tile
        #pragma unroll
        for (int r = 0; r < 8; ++r) {
            if (n == r + mh) s_pos[r + mh] = c[r];  // log2-domain sim_pos
            rowacc[r] += __builtin_amdgcn_exp2f(c[r]);
        }
    } else {                                        // fast path
        #pragma unroll
        for (int r = 0; r < 8; ++r)
            rowacc[r] += __builtin_amdgcn_exp2f(c[r]);
    }
}

// ---------------------------------------------------------------------------
// Kernel 2: fused GEMM (zn @ zn^T, log2-e domain) + exp2 + masked row-sum +
// positive capture. grid.x = 512 row blocks (16 rows); 256 threads = 8 waves.
// Wave w owns column tiles ct = w, w+8, ..., 511 (64 tiles, even count).
// B tiles are explicitly double-buffered (ping-pong) so the 16 b128 loads of
// the next tile fly while the current tile's 8 WMMAs + 8 exps execute.
// ---------------------------------------------------------------------------
__global__ __launch_bounds__(256)
void simclr_main_wmma(const _Float16* __restrict__ zn, float* __restrict__ out) {
    __shared__ float s_denom[16];
    __shared__ float s_pos[16];

    const int tid  = threadIdx.x;
    const int lane = tid & 31;
    const int wave = tid >> 5;
    const int rowbase = blockIdx.x * 16;

    if (tid < 16) { s_denom[tid] = 0.0f; s_pos[tid] = 0.0f; }
    __syncthreads();

    const int h  = lane >> 4;           // lane half
    const int n  = lane & 15;           // tile column (and panel-row selector)
    const int mh = h << 3;              // +0 / +8 on accumulator row index

    // A panel: 16 rows x 256 K, resident in 64 VGPRs for the whole kernel.
    v16h A[8];
    #pragma unroll
    for (int kt = 0; kt < 8; ++kt)
        A[kt] = load_A(zn, rowbase + n, kt * 32, h);

    v8f rowacc = {};                    // per-lane partial row sums

    v16h B0[8], B1[8];
    load_Btile(B0, zn, wave * 16 + n, h);       // prologue: first tile

    for (int ct = wave; ct < 512; ct += 16) {
        // stage tile ct+8 while computing tile ct
        load_Btile(B1, zn, (ct + 8) * 16 + n, h);
        __builtin_prefetch(zn + (size_t)((((ct + 24) & 511) * 16 + n)) * DDIM, 0, 3);
        v8f c = gemm_tile(A, B0);
        epilogue(c, ct * 16, rowbase, n, mh, rowacc, s_pos);

        // stage tile ct+16 (clamped on last iteration) while computing ct+8
        int ctn = (ct + 16 < 512) ? ct + 16 : wave;
        load_Btile(B0, zn, ctn * 16 + n, h);
        c = gemm_tile(A, B1);
        epilogue(c, (ct + 8) * 16, rowbase, n, mh, rowacc, s_pos);
    }

    // Column-reduce each accumulator across the 16 lanes of its half,
    // then merge the 8 waves' partials in LDS.
    #pragma unroll
    for (int r = 0; r < 8; ++r) {
        float v = rowacc[r];
        #pragma unroll
        for (int off = 1; off < 16; off <<= 1)
            v += __shfl_xor(v, off, 32);         // xor<16 stays within each half
        if (n == 0)
            atomicAdd(&s_denom[r + mh], v);      // ds_add_f32
    }
    __syncthreads();

    // loss contribution: sum_m ( ln(denom_m) - ln2 * dot_pos_m ), * 1/(2N)
    if (tid < 16) {
        float v = __logf(s_denom[tid]) - s_pos[tid] * LN2;
        #pragma unroll
        for (int off = 1; off < 16; off <<= 1)
            v += __shfl_xor(v, off, 16);
        if (tid == 0)
            atomicAdd(out, v * (1.0f / TWO_N));  // global_atomic_add_f32
    }
}

// ---------------------------------------------------------------------------
extern "C" void kernel_launch(void* const* d_in, const int* in_sizes, int n_in,
                              void* d_out, int out_size, void* d_ws, size_t ws_size,
                              hipStream_t stream) {
    const float* zi = (const float*)d_in[0];   // [4096, 256] f32
    const float* zj = (const float*)d_in[1];   // [4096, 256] f32
    float* out = (float*)d_out;                // scalar f32 loss
    _Float16* zn = (_Float16*)d_ws;            // [8192, 256] f16 (4 MB scratch)

    simclr_norm_f16<<<TWO_N, 256, 0, stream>>>(zi, zj, zn, out);
    simclr_main_wmma<<<TWO_N / 16, 256, 0, stream>>>(zn, out);
}
